// MPNEncoder_46703474376853
// MI455X (gfx1250) — matrix operand order repaired and checked by reading, not compiled, not run on hardware
//
#include <hip/hip_runtime.h>

#define H      256
#define NMOL   512
#define APM    64
#define BPM    128
#define MAXNB  6
#define AFD    133
#define BFD    147
#define NATOMS (NMOL * APM)   // 32768
#define NBONDS (NMOL * BPM)   // 65536
#define BFPAD  160            // 147 padded to mult of 32
#define CATK   389            // AFD + H
#define CATPAD 416            // padded to mult of 32

typedef __attribute__((ext_vector_type(16))) __bf16       v16bf;
typedef __attribute__((ext_vector_type(8)))  float        v8f;
typedef __attribute__((ext_vector_type(4)))  unsigned int u32x4;
typedef __attribute__((ext_vector_type(4)))  float        f32x4;

union Frag {
  v16bf v;
  u32x4 q[2];
  unsigned short s[16];
};

__device__ __forceinline__ float bf2f(unsigned short u) {
  unsigned int x = ((unsigned int)u) << 16;
  return __builtin_bit_cast(float, x);
}
__device__ __forceinline__ unsigned short f2bf(float f) {
  unsigned int u = __builtin_bit_cast(unsigned int, f);
  u += 0x7FFFu + ((u >> 16) & 1u);   // round-to-nearest-even
  return (unsigned short)(u >> 16);
}

// ---------------------------------------------------------------------------
// Weight repack: W[K x 256] f32 row-major -> per-(kblock, ntile, lane) bf16
// layout for the 16x16x32 bf16 B-fragment in this code's k order: lane (0..31) holds 16
// contiguous K values of column (ntile*16 + (lane&15)), K offset (lane>=16)*16.
// ---------------------------------------------------------------------------
__global__ void pack_weight_kernel(const float* __restrict__ W, int K, int kbc,
                                   unsigned short* __restrict__ out) {
  int total = kbc * 16 * 32 * 16;
  for (int idx = blockIdx.x * blockDim.x + threadIdx.x; idx < total;
       idx += gridDim.x * blockDim.x) {
    int i    = idx & 15;
    int lane = (idx >> 4) & 31;
    int nt   = (idx >> 9) & 15;
    int kb   = idx >> 13;
    int k    = kb * 32 + ((lane >> 4) * 16) + i;
    int col  = nt * 16 + (lane & 15);
    out[idx] = (k < K) ? f2bf(W[k * H + col]) : (unsigned short)0;
  }
}

// Pad+convert bond features: [rows x K] f32 -> [rows x Kpad] bf16 (zero pad)
__global__ void pad_bf16_kernel(const float* __restrict__ in, int K, int Kpad,
                                int rows, unsigned short* __restrict__ out) {
  int total = rows * Kpad;
  for (int idx = blockIdx.x * blockDim.x + threadIdx.x; idx < total;
       idx += gridDim.x * blockDim.x) {
    int r = idx / Kpad, c = idx - r * Kpad;
    out[idx] = (c < K) ? f2bf(in[r * K + c]) : (unsigned short)0;
  }
}

// ---------------------------------------------------------------------------
// Generic GEMM: C[rows x 256] = A_bf16[rows x KBC*32] @ Wpacked, epilogues.
// Block = 256 threads = 8 waves; block computes 16 rows x 256 cols.
// STAGED: packed weights DMA'd into LDS once per block with CDNA5 async
// global->LDS (global_load_async_to_lds_b128 + s_wait_asynccnt), B-fragments
// then come from ds_load_b128. KBC is a template param so the k-loop fully
// unrolls. All A-fragments are preloaded as one load clause so the k-loop is
// just ds_load pairs + back-to-back WMMAs (no per-step global stalls).
// Wave w owns n-tiles {w, w+8}; one A-fragment feeds both WMMAs per k-step.
// The C tile is staged through LDS so the epilogue is fully coalesced.
// mode 1: outB=relu(acc) bf16, outB2=acc bf16 (raw)
// mode 2: outB=relu(acc + addA) bf16
// mode 3: outB=relu(acc) + addA bf16         (attention residual)
// mode 4: outB=relu(acc + 2*addA - addB + bias) bf16   (rxn update)
// mode 5: outF=relu(acc + bias) f32           (final readout)
// ---------------------------------------------------------------------------
template <int KBC, bool STAGED>
__global__ void gemm256_wmma_kernel(const unsigned short* __restrict__ A,
                                    const unsigned short* __restrict__ Wp, int mode,
                                    const unsigned short* __restrict__ addA,
                                    const unsigned short* __restrict__ addB,
                                    const float* __restrict__ bias,
                                    unsigned short* __restrict__ outBf,
                                    unsigned short* __restrict__ outBf2,
                                    float* __restrict__ outF) {
  extern __shared__ unsigned short sW[];   // staged packed weights (dynamic)
  __shared__ float sC[16 * 264];           // C-tile staging, padded stride

  const int tid  = threadIdx.x;
  const int lane = tid & 31;
  const int wave = tid >> 5;
  const int r0   = blockIdx.x * 16;
  constexpr int kpad = KBC * 32;

  if (STAGED) {
    // Async DMA the whole packed weight block (KBC*16KB) into LDS:
    // each iteration is one wave-wide b128 transfer (512B).
    constexpr int chunks = KBC * 1024;  // 16-byte chunks
#pragma unroll
    for (int c = 0; c < chunks / 256; ++c) {
      const int i = c * 256 + tid;
      unsigned loff = (unsigned)(size_t)(const void*)(sW + i * 8);
      unsigned long long ga = (unsigned long long)(const void*)(Wp + i * 8);
      asm volatile("global_load_async_to_lds_b128 %0, %1, off"
                   :: "v"(loff), "v"(ga) : "memory");
    }
    asm volatile("s_wait_asynccnt 0x0" ::: "memory");
    __syncthreads();
  }

  const int arow  = (r0 + (lane & 15)) * kpad;
  const int koff8 = (lane >> 4) * 16;
  const int nt0   = wave;
  const int nt1   = wave + 8;
  const unsigned short* Bsrc = STAGED ? (const unsigned short*)sW : Wp;

  // Preload all A-fragments as one clause; latency hidden by B reads + WMMAs.
  Frag a[KBC];
#pragma unroll
  for (int kb = 0; kb < KBC; ++kb) {
    const unsigned short* ap = A + arow + kb * 32 + koff8;
    a[kb].q[0] = *(const u32x4*)ap;
    a[kb].q[1] = *(const u32x4*)(ap + 8);
  }

  v8f acc0 = {0.f, 0.f, 0.f, 0.f, 0.f, 0.f, 0.f, 0.f};
  v8f acc1 = {0.f, 0.f, 0.f, 0.f, 0.f, 0.f, 0.f, 0.f};

#pragma unroll
  for (int kb = 0; kb < KBC; ++kb) {
    Frag b0, b1;
    const unsigned short* bp0 = Bsrc + (((kb * 16 + nt0) * 32 + lane) << 4);
    const unsigned short* bp1 = Bsrc + (((kb * 16 + nt1) * 32 + lane) << 4);
    b0.q[0] = *(const u32x4*)bp0;
    b0.q[1] = *(const u32x4*)(bp0 + 8);
    b1.q[0] = *(const u32x4*)bp1;
    b1.q[1] = *(const u32x4*)(bp1 + 8);
    acc0 = __builtin_amdgcn_wmma_f32_16x16x32_bf16(false, a[kb].v, false, b0.v,
                                                   (short)0, acc0, false, false);
    acc1 = __builtin_amdgcn_wmma_f32_16x16x32_bf16(false, a[kb].v, false, b1.v,
                                                   (short)0, acc1, false, false);
  }

  // Stage C tile (16 rows x 256 cols f32) into LDS for a coalesced epilogue.
  {
    const int rb = (lane >> 4) * 8;
    const int c0 = nt0 * 16 + (lane & 15);
    const int c1 = nt1 * 16 + (lane & 15);
#pragma unroll
    for (int v = 0; v < 8; ++v) {
      sC[(rb + v) * 264 + c0] = acc0[v];
      sC[(rb + v) * 264 + c1] = acc1[v];
    }
  }
  __syncthreads();

  // Thread t handles row (t>>4), contiguous cols (t&15)*16 .. +15.
  const int erow = tid >> 4;
  const int ecol = (tid & 15) * 16;
  const float* src = sC + erow * 264 + ecol;
  const int base = (r0 + erow) * H + ecol;

  float f[16];
#pragma unroll
  for (int i = 0; i < 16; ++i) f[i] = src[i];

  if (mode == 1) {
    Frag oraw, orelu;
#pragma unroll
    for (int i = 0; i < 16; ++i) {
      oraw.s[i]  = f2bf(f[i]);
      orelu.s[i] = f2bf(f[i] > 0.f ? f[i] : 0.f);
    }
    *(u32x4*)(outBf2 + base)     = oraw.q[0];
    *(u32x4*)(outBf2 + base + 8) = oraw.q[1];
    *(u32x4*)(outBf + base)      = orelu.q[0];
    *(u32x4*)(outBf + base + 8)  = orelu.q[1];
  } else if (mode == 2) {
    Frag add, o;
    add.q[0] = *(const u32x4*)(addA + base);
    add.q[1] = *(const u32x4*)(addA + base + 8);
#pragma unroll
    for (int i = 0; i < 16; ++i) {
      float g = f[i] + bf2f(add.s[i]);
      o.s[i] = f2bf(g > 0.f ? g : 0.f);
    }
    *(u32x4*)(outBf + base)     = o.q[0];
    *(u32x4*)(outBf + base + 8) = o.q[1];
  } else if (mode == 3) {
    Frag add, o;
    add.q[0] = *(const u32x4*)(addA + base);
    add.q[1] = *(const u32x4*)(addA + base + 8);
#pragma unroll
    for (int i = 0; i < 16; ++i) {
      float g = (f[i] > 0.f ? f[i] : 0.f) + bf2f(add.s[i]);
      o.s[i] = f2bf(g);
    }
    *(u32x4*)(outBf + base)     = o.q[0];
    *(u32x4*)(outBf + base + 8) = o.q[1];
  } else if (mode == 4) {
    Frag ra, rb2, o;
    ra.q[0]  = *(const u32x4*)(addA + base);
    ra.q[1]  = *(const u32x4*)(addA + base + 8);
    rb2.q[0] = *(const u32x4*)(addB + base);
    rb2.q[1] = *(const u32x4*)(addB + base + 8);
#pragma unroll
    for (int i = 0; i < 16; ++i) {
      float g = f[i] + 2.f * bf2f(ra.s[i]) - bf2f(rb2.s[i]) + bias[ecol + i];
      o.s[i] = f2bf(g > 0.f ? g : 0.f);
    }
    *(u32x4*)(outBf + base)     = o.q[0];
    *(u32x4*)(outBf + base + 8) = o.q[1];
  } else {  // mode 5
    f32x4 o[4];
#pragma unroll
    for (int i = 0; i < 16; ++i) {
      float g = f[i] + bias[ecol + i];
      o[i >> 2][i & 3] = (g > 0.f ? g : 0.f);
    }
#pragma unroll
    for (int j = 0; j < 4; ++j) *(f32x4*)(outF + base + 4 * j) = o[j];
  }
}

// ---------------------------------------------------------------------------
// Attention: logits[mol][q][j] = scale * sum_h Q[mol,q,h]*S[mol,j,h]
// One molecule per block (8 waves); wave w owns q-tile w, loops 8 j-tiles.
// A-fragments (Q) preloaded once; per j-tile all 8 B-fragments load as one
// clause, then 8 WMMAs fire back-to-back (no per-step load stalls).
// ---------------------------------------------------------------------------
__global__ void attn_logits_kernel(const unsigned short* __restrict__ Q,
                                   const unsigned short* __restrict__ S,
                                   float* __restrict__ Lg) {
  const int mol    = blockIdx.x;
  const int lane   = threadIdx.x & 31;
  const int wave   = threadIdx.x >> 5;
  const int koff8  = (lane >> 4) * 8;
  const int koff16 = (lane >> 4) * 16;
  const int qrow   = (mol * BPM + wave * 16 + (lane & 15)) * H;

  Frag a[8];
#pragma unroll
  for (int kb = 0; kb < 8; ++kb) {
    const unsigned short* ap = Q + qrow + kb * 32 + koff16;
    a[kb].q[0] = *(const u32x4*)ap;
    a[kb].q[1] = *(const u32x4*)(ap + 8);
  }
  const float scale = 0.0625f;  // 1/sqrt(256)
  for (int nt = 0; nt < 8; ++nt) {
    const unsigned short* srow = S + (mol * BPM + nt * 16 + (lane & 15)) * H + koff16;
    Frag b[8];
#pragma unroll
    for (int kb = 0; kb < 8; ++kb) {
      b[kb].q[0] = *(const u32x4*)(srow + kb * 32);
      b[kb].q[1] = *(const u32x4*)(srow + kb * 32 + 8);
    }
    v8f acc = {0.f, 0.f, 0.f, 0.f, 0.f, 0.f, 0.f, 0.f};
#pragma unroll
    for (int kb = 0; kb < 8; ++kb) {
      acc = __builtin_amdgcn_wmma_f32_16x16x32_bf16(false, a[kb].v, false, b[kb].v,
                                                    (short)0, acc, false, false);
    }
    const int colj = nt * 16 + (lane & 15);
    const int base = mol * BPM * BPM;
#pragma unroll
    for (int v = 0; v < 8; ++v) {
      int row = wave * 16 + v + 8 * (lane >> 4);
      Lg[base + row * BPM + colj] = acc[v] * scale;
    }
  }
}

// One wave32 per bond row: coalesced stride-32 reads, shfl reductions.
__global__ void attn_softmax_kernel(const float* __restrict__ Lg,
                                    unsigned short* __restrict__ Ab) {
  const int lane = threadIdx.x & 31;
  const int wave = threadIdx.x >> 5;
  const int row  = blockIdx.x * 8 + wave;
  const float* p = Lg + row * BPM;
  float e[4];
  float mx = -1e30f;
#pragma unroll
  for (int j = 0; j < 4; ++j) {
    e[j] = p[lane + 32 * j];
    mx = fmaxf(mx, e[j]);
  }
#pragma unroll
  for (int m = 16; m >= 1; m >>= 1) mx = fmaxf(mx, __shfl_xor(mx, m, 32));
  float s = 0.f;
#pragma unroll
  for (int j = 0; j < 4; ++j) {
    e[j] = __expf(e[j] - mx);
    s += e[j];
  }
#pragma unroll
  for (int m = 16; m >= 1; m >>= 1) s += __shfl_xor(s, m, 32);
  float inv = 1.f / s;
  unsigned short* o = Ab + row * BPM;
#pragma unroll
  for (int j = 0; j < 4; ++j) o[lane + 32 * j] = f2bf(e[j] * inv);
}

// ST[mol][h][j] = S[mol][j][h]  (so ctx B-fragments are contiguous in K=j)
__global__ void mol_transpose_kernel(const unsigned short* __restrict__ S,
                                     unsigned short* __restrict__ ST) {
  int total = NMOL * H * BPM;
  for (int idx = blockIdx.x * blockDim.x + threadIdx.x; idx < total;
       idx += gridDim.x * blockDim.x) {
    int mol = idx / (H * BPM);
    int rem = idx - mol * (H * BPM);
    int h = rem / BPM, j = rem - h * BPM;
    ST[idx] = S[(mol * BPM + j) * H + h];
  }
}

// ctx[mol][q][h] = sum_j attn[mol][q][j] * S[mol][j][h]
__global__ void attn_ctx_kernel(const unsigned short* __restrict__ Ab,
                                const unsigned short* __restrict__ ST,
                                unsigned short* __restrict__ Ctx) {
  const int mol    = blockIdx.x;
  const int lane   = threadIdx.x & 31;
  const int wave   = threadIdx.x >> 5;
  const int koff8  = (lane >> 4) * 8;
  const int koff16 = (lane >> 4) * 16;

  Frag a[4];  // K = 128
#pragma unroll
  for (int kb = 0; kb < 4; ++kb) {
    const unsigned short* ap =
        Ab + (mol * BPM + wave * 16 + (lane & 15)) * BPM + kb * 32 + koff16;
    a[kb].q[0] = *(const u32x4*)ap;
    a[kb].q[1] = *(const u32x4*)(ap + 8);
  }
  for (int nt = 0; nt < 16; ++nt) {
    const unsigned short* strow = ST + (mol * H + nt * 16 + (lane & 15)) * BPM + koff16;
    Frag b[4];
#pragma unroll
    for (int kb = 0; kb < 4; ++kb) {
      b[kb].q[0] = *(const u32x4*)(strow + kb * 32);
      b[kb].q[1] = *(const u32x4*)(strow + kb * 32 + 8);
    }
    v8f acc = {0.f, 0.f, 0.f, 0.f, 0.f, 0.f, 0.f, 0.f};
#pragma unroll
    for (int kb = 0; kb < 4; ++kb) {
      acc = __builtin_amdgcn_wmma_f32_16x16x32_bf16(false, a[kb].v, false, b[kb].v,
                                                    (short)0, acc, false, false);
    }
    const int h = nt * 16 + (lane & 15);
#pragma unroll
    for (int v = 0; v < 8; ++v) {
      int q = wave * 16 + v + 8 * (lane >> 4);
      Ctx[(mol * BPM + q) * H + h] = f2bf(acc[v]);
    }
  }
}

// ---------------------------------------------------------------------------
// Message aggregation
// ---------------------------------------------------------------------------
__global__ void atom_gather_kernel(const unsigned short* __restrict__ msg,
                                   const int* __restrict__ a2b,
                                   float* __restrict__ amsg) {
  int atom = blockIdx.x, h = threadIdx.x;
  const int* nb = a2b + atom * MAXNB;
  float s = 0.f;
#pragma unroll
  for (int i = 0; i < MAXNB; ++i) s += bf2f(msg[nb[i] * H + h]);
  amsg[atom * H + h] = s;
}

__global__ void bond_combine_kernel(const float* __restrict__ amsg,
                                    const unsigned short* __restrict__ msg,
                                    const int* __restrict__ b2a,
                                    const int* __restrict__ b2revb,
                                    unsigned short* __restrict__ out) {
  int bond = blockIdx.x, h = threadIdx.x;
  int a = b2a[bond], rb = b2revb[bond];
  out[bond * H + h] = f2bf(amsg[a * H + h] - bf2f(msg[rb * H + h]));
}

// xmsg = 2*rmsg - pmsg   (reac_diff)
__global__ void rxn_init_kernel(const unsigned short* __restrict__ r,
                                const unsigned short* __restrict__ p,
                                unsigned short* __restrict__ out) {
  int total = NBONDS * H;
  for (int idx = blockIdx.x * blockDim.x + threadIdx.x; idx < total;
       idx += gridDim.x * blockDim.x)
    out[idx] = f2bf(2.f * bf2f(r[idx]) - bf2f(p[idx]));
}

// cat[atom] = [ 2*rf_atoms - pf_atoms (133) | a_msg (256) | zero pad ]
__global__ void build_cat_kernel(const float* __restrict__ rA,
                                 const float* __restrict__ pA,
                                 const float* __restrict__ amsg,
                                 unsigned short* __restrict__ out) {
  int total = NATOMS * CATPAD;
  for (int idx = blockIdx.x * blockDim.x + threadIdx.x; idx < total;
       idx += gridDim.x * blockDim.x) {
    int atom = idx / CATPAD, c = idx - atom * CATPAD;
    float f;
    if (c < AFD)            f = 2.f * rA[atom * AFD + c] - pA[atom * AFD + c];
    else if (c < AFD + H)   f = amsg[atom * H + (c - AFD)];
    else                    f = 0.f;
    out[idx] = f2bf(f);
  }
}

__global__ void mean_atoms_kernel(const float* __restrict__ hid,
                                  float* __restrict__ out) {
  int mol = blockIdx.x, h = threadIdx.x;
  float s = 0.f;
  for (int a = 0; a < APM; ++a) s += hid[(mol * APM + a) * H + h];
  out[mol * H + h] = s * (1.f / APM);
}

// ---------------------------------------------------------------------------
extern "C" void kernel_launch(void* const* d_in, const int* in_sizes, int n_in,
                              void* d_out, int out_size, void* d_ws, size_t ws_size,
                              hipStream_t stream) {
  (void)in_sizes; (void)n_in; (void)out_size; (void)ws_size;
  const float* rf_atoms = (const float*)d_in[0];
  const float* pf_atoms = (const float*)d_in[1];
  const float* rf_bonds = (const float*)d_in[2];
  const float* pf_bonds = (const float*)d_in[3];
  const int*   a2b      = (const int*)d_in[4];
  const int*   b2a      = (const int*)d_in[5];
  const int*   b2revb   = (const int*)d_in[6];
  const float* W_i      = (const float*)d_in[7];
  const float* W_h      = (const float*)d_in[8];
  const float* W_att    = (const float*)d_in[9];
  const float* W_he     = (const float*)d_in[10];
  const float* b_he     = (const float*)d_in[11];
  const float* W_o      = (const float*)d_in[12];
  const float* b_o      = (const float*)d_in[13];

  char* ws = (char*)d_ws;
  size_t off = 0;
  auto take = [&](size_t bytes) -> char* {
    char* p = ws + off;
    off += (bytes + 255) & ~(size_t)255;
    return p;
  };
  const size_t MSG = (size_t)NBONDS * H * 2;  // bf16 message matrix

  unsigned short* rb_pad = (unsigned short*)take((size_t)NBONDS * BFPAD * 2);
  unsigned short* pb_pad = (unsigned short*)take((size_t)NBONDS * BFPAD * 2);
  unsigned short* rinput = (unsigned short*)take(MSG);
  unsigned short* pinput = (unsigned short*)take(MSG);
  unsigned short* rmsg   = (unsigned short*)take(MSG);
  unsigned short* pmsg   = (unsigned short*)take(MSG);
  unsigned short* rmsg2  = (unsigned short*)take(MSG);
  unsigned short* pmsg2  = (unsigned short*)take(MSG);
  unsigned short* aggout = (unsigned short*)take(MSG);
  unsigned short* ctx    = (unsigned short*)take(MSG);
  unsigned short* STg    = (unsigned short*)take(MSG);
  unsigned short* Ab     = (unsigned short*)take((size_t)NBONDS * BPM * 2);
  float*          Lg     = (float*)take((size_t)NBONDS * BPM * 4);
  float*          amsg   = (float*)take((size_t)NATOMS * H * 4);
  unsigned short* catb   = (unsigned short*)take((size_t)NATOMS * CATPAD * 2);
  float*          hid    = (float*)take((size_t)NATOMS * H * 4);
  unsigned short* pWi    = (unsigned short*)take((size_t)(BFPAD / 32) * 8192 * 2);
  unsigned short* pWh    = (unsigned short*)take((size_t)8 * 8192 * 2);
  unsigned short* pWatt  = (unsigned short*)take((size_t)8 * 8192 * 2);
  unsigned short* pWhe   = (unsigned short*)take((size_t)8 * 8192 * 2);
  unsigned short* pWo    = (unsigned short*)take((size_t)(CATPAD / 32) * 8192 * 2);

  const dim3 B256(256);
  const dim3 Gfill(2048);
  const size_t shW_i = (size_t)(BFPAD / 32) * 16384;  // 80 KB
  const size_t shW_h = (size_t)8 * 16384;             // 128 KB

  // Pack weights (padded K, zero-filled)
  pack_weight_kernel<<<64, B256, 0, stream>>>(W_i, BFD, BFPAD / 32, pWi);
  pack_weight_kernel<<<64, B256, 0, stream>>>(W_h, H, 8, pWh);
  pack_weight_kernel<<<64, B256, 0, stream>>>(W_att, H, 8, pWatt);
  pack_weight_kernel<<<64, B256, 0, stream>>>(W_he, H, 8, pWhe);
  pack_weight_kernel<<<64, B256, 0, stream>>>(W_o, CATK, CATPAD / 32, pWo);

  // Pad/convert bond features
  pad_bf16_kernel<<<Gfill, B256, 0, stream>>>(rf_bonds, BFD, BFPAD, NBONDS, rb_pad);
  pad_bf16_kernel<<<Gfill, B256, 0, stream>>>(pf_bonds, BFD, BFPAD, NBONDS, pb_pad);

  // rinput/pinput = bonds @ W_i ; rmsg/pmsg = relu(.)
  gemm256_wmma_kernel<5, true><<<NBONDS / 16, B256, shW_i, stream>>>(
      rb_pad, pWi, 1, nullptr, nullptr, nullptr, rmsg, rinput, nullptr);
  gemm256_wmma_kernel<5, true><<<NBONDS / 16, B256, shW_i, stream>>>(
      pb_pad, pWi, 1, nullptr, nullptr, nullptr, pmsg, pinput, nullptr);

  for (int it = 0; it < 2; ++it) {
    // rmsg2 = relu(rinput + agg(rmsg) @ W_h)
    atom_gather_kernel<<<NATOMS, B256, 0, stream>>>(rmsg, a2b, amsg);
    bond_combine_kernel<<<NBONDS, B256, 0, stream>>>(amsg, rmsg, b2a, b2revb, aggout);
    gemm256_wmma_kernel<8, true><<<NBONDS / 16, B256, shW_h, stream>>>(
        aggout, pWh, 2, rinput, nullptr, nullptr, rmsg2, nullptr, nullptr);
    // pmsg2 = relu(pinput + agg(pmsg) @ W_h)
    atom_gather_kernel<<<NATOMS, B256, 0, stream>>>(pmsg, a2b, amsg);
    bond_combine_kernel<<<NBONDS, B256, 0, stream>>>(amsg, pmsg, b2a, b2revb, aggout);
    gemm256_wmma_kernel<8, true><<<NBONDS / 16, B256, shW_h, stream>>>(
        aggout, pWh, 2, pinput, nullptr, nullptr, pmsg2, nullptr, nullptr);

    // rmsg = relu(cross_att(S=pmsg2, Q=rmsg2) @ W_att) + rmsg2
    mol_transpose_kernel<<<Gfill, B256, 0, stream>>>(pmsg2, STg);
    attn_logits_kernel<<<NMOL, B256, 0, stream>>>(rmsg2, pmsg2, Lg);
    attn_softmax_kernel<<<NBONDS / 8, B256, 0, stream>>>(Lg, Ab);
    attn_ctx_kernel<<<NMOL, B256, 0, stream>>>(Ab, STg, ctx);
    gemm256_wmma_kernel<8, true><<<NBONDS / 16, B256, shW_h, stream>>>(
        ctx, pWatt, 3, rmsg2, nullptr, nullptr, rmsg, nullptr, nullptr);

    // pmsg = relu(cross_att(S=rmsg2, Q=pmsg2) @ W_att) + pmsg2
    mol_transpose_kernel<<<Gfill, B256, 0, stream>>>(rmsg2, STg);
    attn_logits_kernel<<<NMOL, B256, 0, stream>>>(pmsg2, rmsg2, Lg);
    attn_softmax_kernel<<<NBONDS / 8, B256, 0, stream>>>(Lg, Ab);
    attn_ctx_kernel<<<NMOL, B256, 0, stream>>>(Ab, STg, ctx);
    gemm256_wmma_kernel<8, true><<<NBONDS / 16, B256, shW_h, stream>>>(
        ctx, pWatt, 3, pmsg2, nullptr, nullptr, pmsg, nullptr, nullptr);
  }

  // rxn_msg = 2*rmsg - pmsg ; rxn_input handled in GEMM epilogue (mode 4)
  rxn_init_kernel<<<Gfill, B256, 0, stream>>>(rmsg, pmsg, rmsg2);
  unsigned short* x  = rmsg2;
  unsigned short* xn = pmsg2;
  for (int it = 0; it < 2; ++it) {
    atom_gather_kernel<<<NATOMS, B256, 0, stream>>>(x, a2b, amsg);
    bond_combine_kernel<<<NBONDS, B256, 0, stream>>>(amsg, x, b2a, b2revb, aggout);
    gemm256_wmma_kernel<8, true><<<NBONDS / 16, B256, shW_h, stream>>>(
        aggout, pWhe, 4, rinput, pinput, b_he, xn, nullptr, nullptr);
    unsigned short* t = x; x = xn; xn = t;
  }

  // Readout: a_msg, concat, GEMM with W_o + b_o + relu, mean over atoms
  atom_gather_kernel<<<NATOMS, B256, 0, stream>>>(x, a2b, amsg);
  build_cat_kernel<<<Gfill, B256, 0, stream>>>(rf_atoms, pf_atoms, amsg, catb);
  gemm256_wmma_kernel<13, false><<<NATOMS / 16, B256, 0, stream>>>(
      catb, pWo, 5, nullptr, nullptr, b_o, nullptr, nullptr, hid);
  mean_atoms_kernel<<<NMOL, B256, 0, stream>>>(hid, (float*)d_out);
}
